// MuThought_73950746902590
// MI455X (gfx1250) — compile-verified, matrix operand order checked by reading, not run on hardware
//
#include <hip/hip_runtime.h>
#include <hip/hip_bf16.h>
#include <math.h>

typedef __attribute__((ext_vector_type(16))) _Float16 v16h;
typedef __attribute__((ext_vector_type(8)))  _Float16 v8h;
typedef __attribute__((ext_vector_type(8)))  float    v8f;

// Model dims
#define CD    256      // D
#define CL    64       // L
#define CB    128      // B
#define CNB   4
#define CNH   4
#define CHD   64
#define CMS   6
#define CV    32000
#define CDFF  1024
#define CTOK  8192     // B*L
#define CSTOP 4

// ---------------------------------------------------------------------------
// Block-wide sum over 256 threads
__device__ __forceinline__ float blk_sum256(float v, float* red) {
  int tid = threadIdx.x;
  red[tid] = v; __syncthreads();
  for (int s = 128; s > 0; s >>= 1) {
    if (tid < s) red[tid] += red[tid + s];
    __syncthreads();
  }
  float r = red[0]; __syncthreads();
  return r;
}

// ---------------------------------------------------------------------------
// f32 -> f16 conversion
__global__ void cvt_f16_kernel(const float* __restrict__ X, _Float16* __restrict__ Y, int n) {
  int i = blockIdx.x * 256 + threadIdx.x;
  if (i < n) Y[i] = (_Float16)X[i];
}

// state[row][d] = token_emb[x[row]][d] + pos_emb[row%64][d]
__global__ void embed_kernel(const int* __restrict__ x, const float* __restrict__ te,
                             const float* __restrict__ pe, float* __restrict__ st) {
  int row = blockIdx.x, tid = threadIdx.x;
  int l = row & (CL - 1);
  st[(size_t)row * CD + tid] = te[(size_t)x[row] * CD + tid] + pe[l * CD + tid];
}

// Y[i] = X[i] + vec[i & 255]
__global__ void add_rowvec_kernel(const float* __restrict__ X, const float* __restrict__ vec,
                                  float* __restrict__ Y, int n) {
  int i = blockIdx.x * 256 + threadIdx.x;
  if (i < n) Y[i] = X[i] + vec[i & (CD - 1)];
}

// Y[i] += X[i]
__global__ void add_inplace_kernel(const float* __restrict__ X, float* __restrict__ Y, int n) {
  int i = blockIdx.x * 256 + threadIdx.x;
  if (i < n) Y[i] += X[i];
}

// Row LayerNorm over D=256; one block per row. Y (f16) = LN(X[row]) * w + b
__global__ __launch_bounds__(256) void ln_rows_kernel(const float* __restrict__ X, long rstride,
                                                      const float* __restrict__ w,
                                                      const float* __restrict__ bb,
                                                      _Float16* __restrict__ Y) {
  __shared__ float red[256];
  int row = blockIdx.x, tid = threadIdx.x;
  float xv = X[(size_t)row * rstride + tid];
  float mu = blk_sum256(xv, red) * (1.f / CD);
  float d = xv - mu;
  float var = blk_sum256(d * d, red) * (1.f / CD);
  float y = d * rsqrtf(var + 1e-5f) * w[tid] + bb[tid];
  Y[(size_t)row * CD + tid] = (_Float16)y;
}

// pooled[b] = LN(mean_l state[b,l,:]) ; one block per batch sample
__global__ __launch_bounds__(256) void pool_ln_kernel(const float* __restrict__ S,
                                                      const float* __restrict__ w,
                                                      const float* __restrict__ bb,
                                                      _Float16* __restrict__ Y) {
  __shared__ float red[256];
  int bq = blockIdx.x, tid = threadIdx.x;
  float s = 0.f;
  for (int l = 0; l < CL; ++l) s += S[((size_t)bq * CL + l) * CD + tid];
  s *= (1.f / CL);
  float mu = blk_sum256(s, red) * (1.f / CD);
  float d = s - mu;
  float var = blk_sum256(d * d, red) * (1.f / CD);
  float y = d * rsqrtf(var + 1e-5f) * w[tid] + bb[tid];
  Y[(size_t)bq * CD + tid] = (_Float16)y;
}

// ---------------------------------------------------------------------------
// Issue loads for one 128x32 A tile (async -> LDS) and 32x128 B tile
// (register path, transposed scatter into LDS, n-major with stride 40).
__device__ __forceinline__ void issue_tile_loads(
    const _Float16* __restrict__ A, const _Float16* __restrict__ B,
    int N, int K, int m0, int n0, int kk,
    _Float16* As, _Float16* Bs, int tid) {
  // A tile 128x32: per-lane 16-byte async copies global -> LDS (ASYNCcnt)
#pragma unroll
  for (int i = 0; i < 2; ++i) {
    int linear = (i * 256 + tid) * 8;
    int r = linear >> 5;
    int c = linear & 31;
    unsigned lds = (unsigned)(unsigned long long)&As[r * 40 + c];
    unsigned long long ga = (unsigned long long)&A[(size_t)(m0 + r) * K + (kk + c)];
    asm volatile("global_load_async_to_lds_b128 %0, %1, off"
                 :: "v"(lds), "v"(ga) : "memory");
  }
  // B tile 32x128 -> transposed into Bs[n][k]
#pragma unroll
  for (int i = 0; i < 2; ++i) {
    int linear = (i * 256 + tid) * 8;
    int k = linear >> 7;
    int n = linear & 127;
    v8h bv = *(const v8h*)(&B[(size_t)(kk + k) * N + (n0 + n)]);
    if (kk + 32 < K)
      __builtin_prefetch(&B[(size_t)(kk + 32 + k) * N + (n0 + n)], 0, 1);
#pragma unroll
    for (int j = 0; j < 8; ++j) Bs[(n + j) * 40 + k] = bv[j];
  }
}

// WMMA GEMM: C(MxN,f32) = A(MxK,f16,row) @ B(KxN,f16,row) + bias [, GELU]
// Workgroup: 256 threads = 8 waves; tile 128x128; wave subtile 32x64 (2x4 WMMA).
// Double-buffered LDS; A tiles via GLOBAL_LOAD_ASYNC_TO_LDS_B128.
// Fragment loads follow CDNA5 16-bit 16x32 layout: lane (m=lane&15, half=lane>>4)
// holds K in {8h..8h+7} (elems 0-7) and {16+8h..16+8h+7} (elems 8-15).
__global__ __launch_bounds__(256) void gemm_wmma_f16(
    const _Float16* __restrict__ A, const _Float16* __restrict__ B,
    const float* __restrict__ bias, float* __restrict__ Cf,
    _Float16* __restrict__ Ch, int M, int N, int K, int gelu) {
  __shared__ _Float16 AsBuf[2][128 * 40];
  __shared__ _Float16 BsBuf[2][128 * 40];
  const int tid  = threadIdx.x;
  const int lane = tid & 31;
  const int wave = tid >> 5;
  const int wm = wave & 3;         // wave row 0..3 (32 rows each)
  const int wn = wave >> 2;        // wave col 0..1 (64 cols each)
  const int m0 = blockIdx.y * 128;
  const int n0 = blockIdx.x * 128;
  const int lm = lane & 15;
  const int hf = lane >> 4;

  v8f acc[2][4] = {};

  const int T = K >> 5;  // K / 32 chunks
  issue_tile_loads(A, B, N, K, m0, n0, 0, AsBuf[0], BsBuf[0], tid);

  for (int t = 0; t < T; ++t) {
    _Float16* As = AsBuf[t & 1];
    _Float16* Bs = BsBuf[t & 1];
    // my async copies for tile t complete; barrier publishes all waves' tiles
    asm volatile("s_wait_asynccnt 0x0" ::: "memory");
    __syncthreads();
    // overlap: start tile t+1 into the other buffer while computing tile t
    if (t + 1 < T)
      issue_tile_loads(A, B, N, K, m0, n0, (t + 1) * 32,
                       AsBuf[(t + 1) & 1], BsBuf[(t + 1) & 1], tid);

    v16h afrag[2], bfrag[4];
#pragma unroll
    for (int mi = 0; mi < 2; ++mi) {
      int r = (wm * 32 + mi * 16 + lm) * 40;
      v8h lo = *(const v8h*)(&As[r + 8 * hf]);
      v8h hi = *(const v8h*)(&As[r + 16 + 8 * hf]);
#pragma unroll
      for (int j = 0; j < 8; ++j) { afrag[mi][j] = lo[j]; afrag[mi][j + 8] = hi[j]; }
    }
#pragma unroll
    for (int ni = 0; ni < 4; ++ni) {
      int c = (wn * 64 + ni * 16 + lm) * 40;
      v8h lo = *(const v8h*)(&Bs[c + 8 * hf]);
      v8h hi = *(const v8h*)(&Bs[c + 16 + 8 * hf]);
#pragma unroll
      for (int j = 0; j < 8; ++j) { bfrag[ni][j] = lo[j]; bfrag[ni][j + 8] = hi[j]; }
    }
#pragma unroll
    for (int mi = 0; mi < 2; ++mi)
#pragma unroll
      for (int ni = 0; ni < 4; ++ni)
        acc[mi][ni] = __builtin_amdgcn_wmma_f32_16x16x32_f16(
            false, afrag[mi], false, bfrag[ni], (short)0, acc[mi][ni], false, false);
  }

  // Epilogue: C/D layout — VGPR v holds M = v + 8*half, N = lane&15.
#pragma unroll
  for (int mi = 0; mi < 2; ++mi) {
#pragma unroll
    for (int ni = 0; ni < 4; ++ni) {
      int col = n0 + wn * 64 + ni * 16 + lm;
      float bvv = bias ? bias[col] : 0.f;
#pragma unroll
      for (int v = 0; v < 8; ++v) {
        int row = m0 + wm * 32 + mi * 16 + hf * 8 + v;
        float val = acc[mi][ni][v] + bvv;
        if (gelu) val = 0.5f * val * (1.f + erff(val * 0.70710678118654752f));
        size_t idx = (size_t)row * N + col;
        if (Cf) Cf[idx] = val;
        if (Ch) Ch[idx] = (_Float16)val;
      }
    }
  }
}

// ---------------------------------------------------------------------------
// Fused attention per (b,h): S = softmax(QK^T/8), O = S@V. L=64, HD=64.
__global__ __launch_bounds__(256) void attention_kernel(const _Float16* __restrict__ qkv,
                                                        _Float16* __restrict__ aout) {
  __shared__ _Float16 Qh[64 * 64];
  __shared__ _Float16 Kh[64 * 64];
  __shared__ _Float16 Vh[64 * 64];
  __shared__ float    Sm[64 * 64];
  __shared__ float    red4[64 * 4];
  int bh = blockIdx.x;
  int b = bh >> 2, h = bh & 3;
  int tid = threadIdx.x;
  size_t base = (size_t)b * CL * (3 * CD) + h * CHD;
#pragma unroll 4
  for (int i = 0; i < 16; ++i) {
    int idx = i * 256 + tid;
    int s = idx >> 6, d = idx & 63;
    size_t g = base + (size_t)s * (3 * CD) + d;
    Qh[idx] = qkv[g];
    Kh[idx] = qkv[g + CD];
    Vh[idx] = qkv[g + 2 * CD];
  }
  __syncthreads();
  // scores
#pragma unroll 2
  for (int i = 0; i < 16; ++i) {
    int idx = i * 256 + tid;
    int s = idx >> 6, t = idx & 63;
    float acc = 0.f;
#pragma unroll 8
    for (int d = 0; d < 64; ++d) acc += (float)Qh[s * 64 + d] * (float)Kh[t * 64 + d];
    Sm[idx] = acc * 0.125f;
  }
  __syncthreads();
  // softmax: row r handled by 4 threads (16 cols each)
  int r = tid >> 2, qd = tid & 3;
  float mx = -1e30f;
  for (int j = 0; j < 16; ++j) mx = fmaxf(mx, Sm[r * 64 + qd * 16 + j]);
  red4[r * 4 + qd] = mx; __syncthreads();
  mx = fmaxf(fmaxf(red4[r * 4], red4[r * 4 + 1]), fmaxf(red4[r * 4 + 2], red4[r * 4 + 3]));
  float sm = 0.f;
  for (int j = 0; j < 16; ++j) {
    float e = expf(Sm[r * 64 + qd * 16 + j] - mx);
    Sm[r * 64 + qd * 16 + j] = e;
    sm += e;
  }
  __syncthreads();
  red4[r * 4 + qd] = sm; __syncthreads();
  sm = red4[r * 4] + red4[r * 4 + 1] + red4[r * 4 + 2] + red4[r * 4 + 3];
  float inv = 1.f / sm;
  for (int j = 0; j < 16; ++j) Sm[r * 64 + qd * 16 + j] *= inv;
  __syncthreads();
  // O = S @ V
#pragma unroll 2
  for (int i = 0; i < 16; ++i) {
    int idx = i * 256 + tid;
    int s = idx >> 6, d = idx & 63;
    float acc = 0.f;
#pragma unroll 8
    for (int t = 0; t < 64; ++t) acc += Sm[s * 64 + t] * (float)Vh[t * 64 + d];
    aout[((size_t)b * CL + s) * CD + h * CHD + d] = (_Float16)acc;
  }
}

// ---------------------------------------------------------------------------
// Policy / value heads: plog (B,5) and value (B,) from gelu-activated h1 (f16)
__global__ void heads_kernel(const _Float16* __restrict__ h1p, const _Float16* __restrict__ h1v,
                             const float* __restrict__ p2w, const float* __restrict__ p2b,
                             const float* __restrict__ v2w, const float* __restrict__ v2b,
                             float* __restrict__ plog, float* __restrict__ value) {
  int bq = blockIdx.x, lane = threadIdx.x;
  float acc[5] = {0.f, 0.f, 0.f, 0.f, 0.f};
  float av = 0.f;
  for (int d = lane; d < CD; d += 32) {
    float hp = (float)h1p[(size_t)bq * CD + d];
    float hv = (float)h1v[(size_t)bq * CD + d];
#pragma unroll
    for (int j = 0; j < 5; ++j) acc[j] += hp * p2w[d * 5 + j];
    av += hv * v2w[d];
  }
#pragma unroll
  for (int off = 16; off > 0; off >>= 1) {
#pragma unroll
    for (int j = 0; j < 5; ++j) acc[j] += __shfl_down(acc[j], off);
    av += __shfl_down(av, off);
  }
  if (lane == 0) {
#pragma unroll
    for (int j = 0; j < 5; ++j) plog[bq * 5 + j] = acc[j] + p2b[j];
    value[bq] = av + v2b[0];
  }
}

// Single-block (128 threads == B) routing bookkeeping
__global__ void policy_step_kernel(const float* __restrict__ plog, const float* __restrict__ value,
                                   int* __restrict__ active, int* __restrict__ nsteps,
                                   int* __restrict__ action, int* __restrict__ flags,
                                   float* __restrict__ out_lp, float* __restrict__ out_v,
                                   float* __restrict__ out_ns, float* __restrict__ out_a,
                                   float* __restrict__ out_e, int step) {
  __shared__ int red[128];
  int b = threadIdx.x;
  int act = active[b];
  red[b] = act; __syncthreads();
  for (int s = 64; s > 0; s >>= 1) {
    if (b < s) red[b] |= red[b + s];
    __syncthreads();
  }
  int was_any = red[0]; __syncthreads();

  const float* pl = plog + b * 5;
  int amax = 0; float mx = pl[0];
  for (int j = 1; j < 5; ++j) { if (pl[j] > mx) { mx = pl[j]; amax = j; } }
  float se = 0.f;
  for (int j = 0; j < 5; ++j) se += expf(pl[j] - mx);
  float lse = mx + logf(se);
  float lp = pl[amax] - lse;
  float ent = 0.f;
  for (int j = 0; j < 5; ++j) { float lg = pl[j] - lse; ent -= expf(lg) * lg; }
  float af = act ? 1.f : 0.f;

  out_lp[b * CMS + step] = lp * af;
  out_v[b * CMS + step]  = value[b] * af;
  out_e[b * CMS + step]  = ent * af;
  out_a[b * CMS + step]  = was_any ? (float)amax : (float)CSTOP;
  int ns = nsteps[b] + act;
  nsteps[b] = ns;
  out_ns[b] = (float)ns;
  int newact = act && (amax != CSTOP);
  active[b] = newact;
  action[b] = amax;

  red[b] = newact; __syncthreads();
  for (int s = 64; s > 0; s >>= 1) {
    if (b < s) red[b] |= red[b + s];
    __syncthreads();
  }
  if (b == 0) { flags[0] = was_any; flags[1] = red[0]; }
}

// state[row] = x1[row] + ff2[row] where new_any && action[b]==bl
__global__ void residual_select_kernel(const float* __restrict__ x1, const float* __restrict__ f2,
                                       float* __restrict__ state, const int* __restrict__ action,
                                       const int* __restrict__ flags, int bl) {
  if (!flags[1]) return;
  int row = blockIdx.x;
  if (action[row >> 6] != bl) return;
  int d = threadIdx.x;
  size_t i = (size_t)row * CD + d;
  state[i] = x1[i] + f2[i];
}

__global__ void init_flags_kernel(int* __restrict__ active, int* __restrict__ nsteps,
                                  int* __restrict__ flags) {
  int b = threadIdx.x;
  active[b] = 1;
  nsteps[b] = 0;
  if (b == 0) { flags[0] = 1; flags[1] = 1; }
}

// ---------------------------------------------------------------------------
extern "C" void kernel_launch(void* const* d_in, const int* in_sizes, int n_in,
                              void* d_out, int out_size, void* d_ws, size_t ws_size,
                              hipStream_t stream) {
  (void)in_sizes; (void)n_in; (void)out_size; (void)ws_size;
  const int*   x       = (const int*)  d_in[0];
  const float* tokemb  = (const float*)d_in[1];
  const float* posemb  = (const float*)d_in[2];
  const float* stepemb = (const float*)d_in[3];
  const float* blkemb  = (const float*)d_in[4];
  const float* n1w = (const float*)d_in[5],  *n1b = (const float*)d_in[6];
  const float* qkvw = (const float*)d_in[7], *qkvb = (const float*)d_in[8];
  const float* aow = (const float*)d_in[9],  *aob = (const float*)d_in[10];
  const float* n2w = (const float*)d_in[11], *n2b = (const float*)d_in[12];
  const float* f1w = (const float*)d_in[13], *f1b = (const float*)d_in[14];
  const float* f2w = (const float*)d_in[15], *f2b = (const float*)d_in[16];
  const float* nw  = (const float*)d_in[17], *nb  = (const float*)d_in[18];
  const float* p1w = (const float*)d_in[19], *p1b = (const float*)d_in[20];
  const float* p2w = (const float*)d_in[21], *p2b = (const float*)d_in[22];
  const float* v1w = (const float*)d_in[23], *v1b = (const float*)d_in[24];
  const float* v2w = (const float*)d_in[25], *v2b = (const float*)d_in[26];
  const float* decw = (const float*)d_in[27], *decb = (const float*)d_in[28];
  float* out = (float*)d_out;

  // ---- workspace carve (256B aligned) ----
  size_t off = 0;
  char* wsb = (char*)d_ws;
  auto take = [&](size_t bytes) -> void* {
    void* p = wsb + off;
    off += (bytes + 255) & ~(size_t)255;
    return p;
  };
  _Float16* w16_qkv = (_Float16*)take((size_t)CNB * CD * 3 * CD * 2);
  _Float16* w16_ao  = (_Float16*)take((size_t)CNB * CD * CD * 2);
  _Float16* w16_ff1 = (_Float16*)take((size_t)CNB * CD * CDFF * 2);
  _Float16* w16_ff2 = (_Float16*)take((size_t)CNB * CDFF * CD * 2);
  _Float16* w16_p1  = (_Float16*)take((size_t)CD * CD * 2);
  _Float16* w16_v1  = (_Float16*)take((size_t)CD * CD * 2);
  _Float16* w16_dec = (_Float16*)take((size_t)CD * CV * 2);
  float*    state   = (float*)take((size_t)CTOK * CD * 4);
  float*    s_step  = (float*)take((size_t)CTOK * CD * 4);
  float*    x0      = (float*)take((size_t)CTOK * CD * 4);     // becomes x1 after residual
  float*    proj    = (float*)take((size_t)CTOK * CD * 4);     // proj / ff2 scratch
  _Float16* ln16    = (_Float16*)take((size_t)CTOK * CD * 2);
  _Float16* qkv16   = (_Float16*)take((size_t)CTOK * 3 * CD * 2);
  _Float16* attn16  = (_Float16*)take((size_t)CTOK * CD * 2);
  _Float16* ffh16   = (_Float16*)take((size_t)CTOK * CDFF * 2);
  _Float16* pooled16= (_Float16*)take((size_t)CB * CD * 2);
  _Float16* h1p16   = (_Float16*)take((size_t)CB * CD * 2);
  _Float16* h1v16   = (_Float16*)take((size_t)CB * CD * 2);
  float*    plog    = (float*)take((size_t)CB * 5 * 4);
  float*    value   = (float*)take((size_t)CB * 4);
  int*      active  = (int*)take((size_t)CB * 4);
  int*      nsteps  = (int*)take((size_t)CB * 4);
  int*      action  = (int*)take((size_t)CB * 4);
  int*      flags   = (int*)take(2 * 4);

  // ---- output layout (tuple flattened) ----
  float* out_base  = out;
  float* out_final = out + (size_t)CB * CV;
  float* out_lp    = out + 2 * (size_t)CB * CV;
  float* out_v     = out_lp + CB * CMS;
  float* out_ns    = out_v + CB * CMS;
  float* out_a     = out_ns + CB;
  float* out_e     = out_a + CB * CMS;

  auto cvt = [&](const float* src, _Float16* dst, int n) {
    cvt_f16_kernel<<<(n + 255) / 256, 256, 0, stream>>>(src, dst, n);
  };
  auto gemm = [&](const _Float16* A, const _Float16* Bm, const float* bias,
                  float* Cf, _Float16* Ch, int M, int N, int K, int gelu) {
    dim3 g(N / 128, M / 128);
    gemm_wmma_f16<<<g, 256, 0, stream>>>(A, Bm, bias, Cf, Ch, M, N, K, gelu);
  };

  // ---- weight conversion to f16 ----
  cvt(qkvw, w16_qkv, CNB * CD * 3 * CD);
  cvt(aow,  w16_ao,  CNB * CD * CD);
  cvt(f1w,  w16_ff1, CNB * CD * CDFF);
  cvt(f2w,  w16_ff2, CNB * CDFF * CD);
  cvt(p1w,  w16_p1,  CD * CD);
  cvt(v1w,  w16_v1,  CD * CD);
  cvt(decw, w16_dec, CD * CV);

  const int NTD = CTOK * CD;  // 2M elements

  // ---- encode + base decode ----
  embed_kernel<<<CTOK, CD, 0, stream>>>(x, tokemb, posemb, state);
  pool_ln_kernel<<<CB, CD, 0, stream>>>(state, nw, nb, pooled16);
  gemm(pooled16, w16_dec, decb, out_base, nullptr, CB, CV, CD, 0);
  init_flags_kernel<<<1, CB, 0, stream>>>(active, nsteps, flags);

  // ---- thinking steps ----
  for (int step = 0; step < CMS; ++step) {
    // policy/value from last token (pre-update state)
    ln_rows_kernel<<<CB, CD, 0, stream>>>(state + (size_t)(CL - 1) * CD, (long)CL * CD,
                                          nw, nb, pooled16);
    gemm(pooled16, w16_p1, p1b, nullptr, h1p16, CB, CD, CD, 1);
    gemm(pooled16, w16_v1, v1b, nullptr, h1v16, CB, CD, CD, 1);
    heads_kernel<<<CB, 32, 0, stream>>>(h1p16, h1v16, p2w, p2b, v2w, v2b, plog, value);
    policy_step_kernel<<<1, CB, 0, stream>>>(plog, value, active, nsteps, action, flags,
                                             out_lp, out_v, out_ns, out_a, out_e, step);

    // s_step = state + step_emb[step]
    add_rowvec_kernel<<<NTD / 256, 256, 0, stream>>>(state, stepemb + step * CD, s_step, NTD);

    for (int bl = 0; bl < CNB; ++bl) {
      // x0 = s_step + block_emb[bl]
      add_rowvec_kernel<<<NTD / 256, 256, 0, stream>>>(s_step, blkemb + bl * CD, x0, NTD);
      // attn branch
      ln_rows_kernel<<<CTOK, CD, 0, stream>>>(x0, CD, n1w + bl * CD, n1b + bl * CD, ln16);
      gemm(ln16, w16_qkv + (size_t)bl * CD * 3 * CD, qkvb + bl * 3 * CD,
           nullptr, qkv16, CTOK, 3 * CD, CD, 0);
      attention_kernel<<<CB * CNH, 256, 0, stream>>>(qkv16, attn16);
      gemm(attn16, w16_ao + (size_t)bl * CD * CD, aob + bl * CD,
           proj, nullptr, CTOK, CD, CD, 0);
      add_inplace_kernel<<<NTD / 256, 256, 0, stream>>>(proj, x0, NTD);  // x0 -> x1
      // ffn branch
      ln_rows_kernel<<<CTOK, CD, 0, stream>>>(x0, CD, n2w + bl * CD, n2b + bl * CD, ln16);
      gemm(ln16, w16_ff1 + (size_t)bl * CD * CDFF, f1b + bl * CDFF,
           nullptr, ffh16, CTOK, CDFF, CD, 1);
      gemm(ffh16, w16_ff2 + (size_t)bl * CDFF * CD, f2b + bl * CD,
           proj, nullptr, CTOK, CD, CDFF, 0);
      // state[row] = x1 + ff2 for rows routed to this block (gated by new_any)
      residual_select_kernel<<<CTOK, CD, 0, stream>>>(x0, proj, state, action, flags, bl);
    }
  }

  // ---- final decode ----
  pool_ln_kernel<<<CB, CD, 0, stream>>>(state, nw, nb, pooled16);
  gemm(pooled16, w16_dec, decb, out_final, nullptr, CB, CV, CD, 0);
}
